// QKVDSEDemoBlock_12850542149907
// MI455X (gfx1250) — matrix operand order repaired and checked by reading, not compiled, not run on hardware
//
#include <hip/hip_runtime.h>

// ---------------------------------------------------------------------------
// Fused: probs = softmax-ish(exp(x+b1)); ctx = probs@M1; hidden = ctx + x;
//        out  = (hidden+b2)@M2 + hidden.       fp32 in/out, f16 WMMA GEMMs.
// ---------------------------------------------------------------------------

typedef _Float16 v16h __attribute__((ext_vector_type(16)));
typedef float    v8f  __attribute__((ext_vector_type(8)));

#define WAVES_PER_WG 8
#define TILE_ITERS   8

static __device__ __forceinline__ void wave_sync() {
  // LDS ops from one wave are executed in order by hardware; these only stop
  // the compiler from reordering cross-lane LDS communication.
  __builtin_amdgcn_fence(__ATOMIC_ACQ_REL, "wavefront");
  __builtin_amdgcn_wave_barrier();
  __builtin_amdgcn_fence(__ATOMIC_ACQ_REL, "wavefront");
}

static __device__ __forceinline__ v8f wmma_f16(v16h a, v16h b, v8f c) {
  // (neg_a, A, neg_b, B, c_mod, C, reuse_a, reuse_b)
  return __builtin_amdgcn_wmma_f32_16x16x32_f16(false, a, false, b,
                                                (short)0, c, false, false);
}

__global__ __launch_bounds__(256) void fused_mix_kernel(
    const float* __restrict__ x, const float* __restrict__ mix1,
    const float* __restrict__ mix2, const float* __restrict__ bias1,
    const float* __restrict__ bias2, float* __restrict__ out, long n_rows) {
  // B fragments in WMMA layout: [matrix][n_tile][k_tile][lane][elem] (16 KB)
  __shared__ alignas(32) _Float16 Bfrag[2][4][2][32][16];
  // Per-wave 16x64 f32 staging tile (ctx -> hidden -> out), 8 x 4 KB
  __shared__ alignas(16) float tileBuf[WAVES_PER_WG][16 * 64];

  const int tid  = threadIdx.x;
  const int wave = tid >> 5;
  const int lane = tid & 31;
  const int r    = lane & 15;  // row within 16-row tile (A layout); column in C layout
  const int g    = lane >> 4;  // lane half

  // ---- Build B fragments once per workgroup (cooperative, 512 lane-slots).
  // B (K x 16) layout: lane l holds column N = l%16; element i -> K = 8*(l/16) + 16*(i/8) + i%8.
  {
    const float* mixes[2] = {mix1, mix2};
    for (int s = tid; s < 512; s += 256) {
      const int m  = s >> 8;
      const int rem = s & 255;
      const int nt = rem >> 6;
      const int kt = (rem >> 5) & 1;
      const int lv = rem & 31;
      const int Nn = 16 * nt + (lv & 15);
      const int gg = lv >> 4;
      const float* Mx = mixes[m];
#pragma unroll
      for (int i = 0; i < 16; ++i) {
        const int K = 32 * kt + 8 * gg + 16 * (i >> 3) + (i & 7);
        Bfrag[m][nt][kt][lv][i] = (_Float16)Mx[K * 64 + Nn];
      }
    }
  }

  // ---- Per-lane bias registers at this lane's A-layout columns: 8g + 16t + j
  float b1v[4][8], b2v[4][8];
#pragma unroll
  for (int t = 0; t < 4; ++t) {
#pragma unroll
    for (int j = 0; j < 8; ++j) {
      const int c = 16 * t + 8 * g + j;
      b1v[t][j] = bias1[c];
      b2v[t][j] = bias2[c];
    }
  }

  __syncthreads();  // Bfrag visible to all waves

  float* tb = tileBuf[wave];

  for (int it = 0; it < TILE_ITERS; ++it) {
    const long tile =
        (long)blockIdx.x * (WAVES_PER_WG * TILE_ITERS) + it * WAVES_PER_WG + wave;
    const long r0 = tile * 16;
    const bool active = (r0 < n_rows);  // wave-uniform

    // ---- Load x tile: lane holds row r, cols {8g+16t .. 8g+16t+7}, t=0..3
    float xv[4][8];
    if (active) {
      const float* xr = x + (size_t)(r0 + r) * 64;
#pragma unroll
      for (int t = 0; t < 4; ++t) {
        const float4 lo = *(const float4*)(xr + 16 * t + 8 * g);
        const float4 hi = *(const float4*)(xr + 16 * t + 8 * g + 4);
        xv[t][0] = lo.x; xv[t][1] = lo.y; xv[t][2] = lo.z; xv[t][3] = lo.w;
        xv[t][4] = hi.x; xv[t][5] = hi.y; xv[t][6] = hi.z; xv[t][7] = hi.w;
      }
      // prefetch next tile's row for this wave (global_prefetch_b8)
      if (r0 + WAVES_PER_WG * 16 + r < n_rows)
        __builtin_prefetch(x + (size_t)(r0 + WAVES_PER_WG * 16 + r) * 64, 0, 0);
    } else {
#pragma unroll
      for (int t = 0; t < 4; ++t)
#pragma unroll
        for (int j = 0; j < 8; ++j) xv[t][j] = 0.0f;
    }

    // ---- probs = exp(x+b1) / rowsum  (no max-subtract, matching reference)
    float ev[4][8];
    float s = 0.0f;
#pragma unroll
    for (int t = 0; t < 4; ++t) {
#pragma unroll
      for (int j = 0; j < 8; ++j) {
        const float e = __expf(xv[t][j] + b1v[t][j]);
        ev[t][j] = e;
        s += e;
      }
    }
    s += __shfl_xor(s, 16);  // combine the two lane halves of each row
    const float rs = 1.0f / s;

    // ---- Pack A fragments (16x32 f16 layout, §7.12.2): frag kt <- chunks 2kt,2kt+1
    v16h a0, a1;
#pragma unroll
    for (int i = 0; i < 16; ++i) {
      a0[i] = (_Float16)(ev[(i >> 3)][i & 7] * rs);
      a1[i] = (_Float16)(ev[2 + (i >> 3)][i & 7] * rs);
    }

    // ---- GEMM1: ctx = probs @ mix1 -> write to LDS tile in C/D layout
#pragma unroll
    for (int nt = 0; nt < 4; ++nt) {
      v8f acc = {0.f, 0.f, 0.f, 0.f, 0.f, 0.f, 0.f, 0.f};
      const v16h bA = *(const v16h*)&Bfrag[0][nt][0][lane][0];
      acc = wmma_f16(a0, bA, acc);
      const v16h bB = *(const v16h*)&Bfrag[0][nt][1][lane][0];
      acc = wmma_f16(a1, bB, acc);
#pragma unroll
      for (int v = 0; v < 8; ++v)
        tb[(v + 8 * g) * 64 + 16 * nt + r] = acc[v];  // M=v+8g, N=16nt+r
    }

    wave_sync();  // C-layout writes -> A-layout reads

    // ---- hidden = ctx + x; store hidden back; build A2 = f16(hidden + b2)
    float hv[4][8];
#pragma unroll
    for (int t = 0; t < 4; ++t) {
      float* hp = tb + r * 64 + 16 * t + 8 * g;
      const float4 lo = *(const float4*)(hp);
      const float4 hi = *(const float4*)(hp + 4);
      hv[t][0] = lo.x + xv[t][0]; hv[t][1] = lo.y + xv[t][1];
      hv[t][2] = lo.z + xv[t][2]; hv[t][3] = lo.w + xv[t][3];
      hv[t][4] = hi.x + xv[t][4]; hv[t][5] = hi.y + xv[t][5];
      hv[t][6] = hi.z + xv[t][6]; hv[t][7] = hi.w + xv[t][7];
      float4 wlo, whi;
      wlo.x = hv[t][0]; wlo.y = hv[t][1]; wlo.z = hv[t][2]; wlo.w = hv[t][3];
      whi.x = hv[t][4]; whi.y = hv[t][5]; whi.z = hv[t][6]; whi.w = hv[t][7];
      *(float4*)(hp) = wlo;
      *(float4*)(hp + 4) = whi;
    }

    v16h a20, a21;
#pragma unroll
    for (int i = 0; i < 16; ++i) {
      const int t0 = (i >> 3), t1 = 2 + (i >> 3), j = i & 7;
      a20[i] = (_Float16)(hv[t0][j] + b2v[t0][j]);
      a21[i] = (_Float16)(hv[t1][j] + b2v[t1][j]);
    }

    wave_sync();  // A-layout hidden writes -> C-layout reads

    // ---- GEMM2: out = (hidden+b2) @ mix2 + hidden  (RMW the LDS tile)
#pragma unroll
    for (int nt = 0; nt < 4; ++nt) {
      v8f acc = {0.f, 0.f, 0.f, 0.f, 0.f, 0.f, 0.f, 0.f};
      const v16h bA = *(const v16h*)&Bfrag[1][nt][0][lane][0];
      acc = wmma_f16(a20, bA, acc);
      const v16h bB = *(const v16h*)&Bfrag[1][nt][1][lane][0];
      acc = wmma_f16(a21, bB, acc);
#pragma unroll
      for (int v = 0; v < 8; ++v) {
        const int idx = (v + 8 * g) * 64 + 16 * nt + r;
        tb[idx] = acc[v] + tb[idx];
      }
    }

    wave_sync();  // out writes -> contiguous readback

    // ---- Coalesced store: 8 x global_store_b128, 512B contiguous per instr
    if (active) {
      float* op = out + (size_t)r0 * 64;
#pragma unroll
      for (int q = 0; q < 8; ++q) {
        const float4 vdat = *(const float4*)&tb[q * 128 + lane * 4];
        *(float4*)(op + q * 128 + lane * 4) = vdat;
      }
    }

    wave_sync();  // readback done before next iteration reuses the tile
  }
}

extern "C" void kernel_launch(void* const* d_in, const int* in_sizes, int n_in,
                              void* d_out, int out_size, void* d_ws, size_t ws_size,
                              hipStream_t stream) {
  (void)n_in; (void)out_size; (void)d_ws; (void)ws_size;
  const float* x  = (const float*)d_in[0];
  const float* m1 = (const float*)d_in[1];
  const float* m2 = (const float*)d_in[2];
  const float* b1 = (const float*)d_in[3];
  const float* b2 = (const float*)d_in[4];
  float* o = (float*)d_out;

  const long n_rows = (long)in_sizes[0] / 64;                 // N
  const long tiles = (n_rows + 15) / 16;                      // 16-row tiles
  const long tiles_per_wg = (long)WAVES_PER_WG * TILE_ITERS;  // 64
  const int grid = (int)((tiles + tiles_per_wg - 1) / tiles_per_wg);

  fused_mix_kernel<<<grid, 256, 0, stream>>>(x, m1, m2, b1, b2, o, n_rows);
}
